// Attention_77043123355775
// MI455X (gfx1250) — matrix operand order repaired and checked by reading, not compiled, not run on hardware
//
#include <hip/hip_runtime.h>
#include <stdint.h>

#define BATCH 2
#define SEQ   2048
#define DIMSZ 2048
#define NH    16
#define NKV   8
#define HD    128
#define MROWS (BATCH*SEQ)   // 4096

typedef __attribute__((ext_vector_type(16))) _Float16 v16h;
typedef __attribute__((ext_vector_type(4)))  _Float16 h4;
typedef __attribute__((ext_vector_type(8)))  float    v8f;
typedef __attribute__((ext_vector_type(4)))  float    f4;

// ---- fragment loaders (16B + 16B, or 32B contiguous) ----
__device__ static inline v16h ld_2x16(const _Float16* p0, const _Float16* p1) {
  union { v16h v; f4 q[2]; } u;
  u.q[0] = *(const f4*)p0;
  u.q[1] = *(const f4*)p1;
  return u.v;
}
__device__ static inline v16h ld_32(const _Float16* p) {
  union { v16h v; f4 q[2]; } u;
  u.q[0] = *(const f4*)p;
  u.q[1] = *(const f4*)(p + 8);
  return u.v;
}

// ---- CDNA5 async global->LDS copy (16B per lane), GVS addressing ----
__device__ static inline void async_cp16(unsigned lds_addr, const void* base, int goff) {
  asm volatile("global_load_async_to_lds_b128 %0, %1, %2"
               :: "v"(lds_addr), "v"(goff), "s"(base) : "memory");
}
template <int N>
__device__ static inline void wait_async() {
  asm volatile("s_wait_asynccnt %0" :: "i"(N) : "memory");
}

// ================= fp32 -> fp16 conversion (vectorized x4) =================
__global__ void cvt_f32_f16v4(const float* __restrict__ in, _Float16* __restrict__ out, int n4) {
  int i = blockIdx.x * blockDim.x + threadIdx.x;
  if (i < n4) {
    f4 v = ((const f4*)in)[i];
    h4 r = { (_Float16)v[0], (_Float16)v[1], (_Float16)v[2], (_Float16)v[3] };
    *(h4*)(out + 4 * (size_t)i) = r;
  }
}

// ================= tiled WMMA GEMM: C[M,N] = A[M,K] * W[N,K]^T =================
// K-step 64: 16 WMMAs per barrier pair, double-buffered async LDS staging.
// MODE 0: Q proj + RoPE  -> Qbuf  [b][h(16)][s][d]      (fp16)
// MODE 1: K proj + RoPE  -> Kbuf  [b][kvh(8)][s][d]     (fp16)
// MODE 2: V proj         -> VTbuf [b][kvh(8)][d][s]     (fp16, transposed)
// MODE 3: O proj         -> out   [M][N] fp32 (d_out)
template <int MODE>
__global__ __launch_bounds__(256) void gemm_wmma(
    const _Float16* __restrict__ A, const _Float16* __restrict__ W,
    void* __restrict__ outp, int M, int N, int K)
{
  __shared__ __align__(16) _Float16 tA[2][128 * 64];
  __shared__ __align__(16) _Float16 tB[2][128 * 64];

  const int t    = threadIdx.x;
  const int lane = t & 31;
  const int wave = t >> 5;
  const int lh   = lane >> 4;       // lane half (0/1)
  const int l16  = lane & 15;
  const int mbase = blockIdx.y * 128;
  const int nbase = blockIdx.x * 128;
  const int wm = (wave >> 2) * 64;  // wave row offset
  const int wn = (wave & 3) * 32;   // wave col offset

  v8f acc[4][2];
#pragma unroll
  for (int i = 0; i < 4; i++)
#pragma unroll
    for (int j = 0; j < 2; j++)
      acc[i][j] = (v8f){0.f,0.f,0.f,0.f,0.f,0.f,0.f,0.f};

  // stage 128x64 fp16 tiles (16KB each): 1024 x 16B chunks, 4 per thread per matrix
  auto stage = [&](int k0, int buf) {
    const unsigned ldsA = (unsigned)(uintptr_t)&tA[buf][0];
    const unsigned ldsB = (unsigned)(uintptr_t)&tB[buf][0];
#pragma unroll
    for (int i = 0; i < 4; i++) {
      const int c = t + 256 * i;          // 0..1023
      const int r = c >> 3;               // row (8 chunks per 128B row)
      const int o = (c & 7) * 16;         // byte offset within row
      async_cp16(ldsA + c * 16, A, ((mbase + r) * K + k0) * 2 + o);
      async_cp16(ldsB + c * 16, W, ((nbase + r) * K + k0) * 2 + o);
    }
  };

  const int nk = K / 64;
  stage(0, 0);
  for (int kt = 0; kt < nk; kt++) {
    const int cur = kt & 1;
    if (kt + 1 < nk) { stage((kt + 1) * 64, 1 - cur); wait_async<8>(); }
    else             { wait_async<0>(); }
    __syncthreads();

#pragma unroll
    for (int kk = 0; kk < 64; kk += 32) {
      v16h bf[2], af[4];
#pragma unroll
      for (int j = 0; j < 2; j++)
        bf[j] = ld_32(&tB[cur][(wn + j * 16 + l16) * 64 + kk + lh * 16]);
#pragma unroll
      for (int i = 0; i < 4; i++) {
        const _Float16* pr = &tA[cur][(wm + i * 16 + l16) * 64 + kk];
        af[i] = ld_32(pr + lh * 16);
      }
#pragma unroll
      for (int i = 0; i < 4; i++)
#pragma unroll
        for (int j = 0; j < 2; j++)
          acc[i][j] = __builtin_amdgcn_wmma_f32_16x16x32_f16(
              false, af[i], false, bf[j], (short)0, acc[i][j], false, false);
    }
    __syncthreads();  // reads done before next stage overwrites this buffer
  }

  // ---------------- epilogue ----------------
#pragma unroll
  for (int i = 0; i < 4; i++) {
#pragma unroll
    for (int j = 0; j < 2; j++) {
      union { v8f v; float f[8]; } u; u.v = acc[i][j];
      const int n = nbase + wn + j * 16 + l16;
      if (MODE == 3) {
        float* out = (float*)outp;
#pragma unroll
        for (int r = 0; r < 8; r++) {
          const int m = mbase + wm + i * 16 + r + lh * 8;
          out[(size_t)m * N + n] = u.f[r];
        }
      } else if (MODE == 2) {
        _Float16* out = (_Float16*)outp;
        const int d = n & 127, h = n >> 7;
#pragma unroll
        for (int r = 0; r < 8; r++) {
          const int m = mbase + wm + i * 16 + r + lh * 8;
          const int s = m & (SEQ - 1), b = m >> 11;
          out[(((size_t)(b * NKV + h) * HD + d) * SEQ) + s] = (_Float16)u.f[r];
        }
      } else { // MODE 0/1: fused RoPE
        _Float16* out = (_Float16*)outp;
        const int d = n & 127, h = n >> 7;
        const int p = d >> 1;
        const bool odd = (d & 1);
        const float inv = __expf(-0.143911568f * (float)p); // 10000^(-2p/128)
#pragma unroll
        for (int r = 0; r < 8; r++) {
          const int m = mbase + wm + i * 16 + r + lh * 8;
          const int s = m & (SEQ - 1), b = m >> 11;
          float sn, cs;
          __sincosf((float)s * inv, &sn, &cs);
          const float mine  = u.f[r];
          const float other = __shfl_xor(mine, 1, 32); // even/odd cols adjacent lanes
          const float val = odd ? (other * sn + mine * cs)
                                : (mine * cs - other * sn);
          const int nheads = (MODE == 0) ? NH : NKV;
          out[(((size_t)(b * nheads + h) * SEQ) + s) * HD + d] = (_Float16)val;
        }
      }
    }
  }
}

// ========== causal flash attention: block = 4 waves = 64 q-rows, shared K/V LDS tiles ==========
__global__ __launch_bounds__(128) void flash_wmma(
    const _Float16* __restrict__ Q, const _Float16* __restrict__ Kc,
    const _Float16* __restrict__ VT, _Float16* __restrict__ attn)
{
  __shared__ __align__(16) _Float16 ktile[2][32 * 128];  // [key][d]
  __shared__ __align__(16) _Float16 vtile[2][128 * 32];  // [d][key]
  __shared__ __align__(16) _Float16 pbuf[4][16 * 32];

  const int t    = threadIdx.x;
  const int lane = t & 31;
  const int wave = t >> 5;
  const int lh   = lane >> 4;
  const int l16  = lane & 15;

  const int qb  = (blockIdx.x * 4 + wave) * 16;
  const int bh  = blockIdx.y;
  const int b   = bh >> 4, h = bh & 15, kvh = h >> 1;

  const _Float16* Qp = Q  + ((size_t)(b * NH  + h)   * SEQ) * HD;
  const _Float16* Kp = Kc + ((size_t)(b * NKV + kvh) * SEQ) * HD;
  const _Float16* Vp = VT + ((size_t)(b * NKV + kvh) * HD) * SEQ;

  auto stage_kv = [&](int kb, int buf) {
    const unsigned ldsK = (unsigned)(uintptr_t)&ktile[buf][0];
    const unsigned ldsV = (unsigned)(uintptr_t)&vtile[buf][0];
#pragma unroll
    for (int i = 0; i < 4; i++) {
      const int c = t + 128 * i;            // 0..511, 16B chunks
      async_cp16(ldsK + c * 16, Kp, ((kb + (c >> 4)) * HD) * 2 + (c & 15) * 16);
      async_cp16(ldsV + c * 16, Vp, ((c >> 2) * SEQ + kb) * 2 + (c & 3) * 16);
    }
  };

  // Q fragments (A-layout): 4 chunks of head-dim 32
  v16h qf[4];
#pragma unroll
  for (int c = 0; c < 4; c++) {
    const _Float16* pr = Qp + (size_t)(qb + l16) * HD + c * 32;
    qf[c] = ld_32(pr + lh * 16);
  }

  // all-ones B fragment: rowsum(P) = P x ones via one extra WMMA
  union { v16h v; _Float16 hh[16]; } u1;
#pragma unroll
  for (int k = 0; k < 16; k++) u1.hh[k] = (_Float16)1.0f;
  const v16h ones = u1.v;

  v8f o[8];
#pragma unroll
  for (int fn = 0; fn < 8; fn++) o[fn] = (v8f){0.f,0.f,0.f,0.f,0.f,0.f,0.f,0.f};
  float mrun[8], lrun[8];
#pragma unroll
  for (int r = 0; r < 8; r++) { mrun[r] = -1e30f; lrun[r] = 0.f; }
  const float scale = 0.08838834764f; // 1/sqrt(128)

  // uniform trip count across the block (barriers outside compute branch)
  const int ntiles = blockIdx.x * 2 + 2;
  stage_kv(0, 0);
  for (int it = 0; it < ntiles; it++) {
    const int kb  = it * 32;
    const int cur = it & 1;
    if (it + 1 < ntiles) { stage_kv((it + 1) * 32, 1 - cur); wait_async<8>(); }
    else                 { wait_async<0>(); }
    __syncthreads();

    if (kb <= qb + 15) {   // wave-uniform: skip fully-masked tiles
      // ---- scores: load all 8 K fragments, then 8 WMMAs ----
      v16h kf0[4], kf1[4];
#pragma unroll
      for (int c = 0; c < 4; c++) {
        kf0[c] = ld_32(&ktile[cur][(     l16) * HD + c * 32 + lh * 16]);
        kf1[c] = ld_32(&ktile[cur][(16 + l16) * HD + c * 32 + lh * 16]);
      }
      v8f s0 = (v8f){0.f,0.f,0.f,0.f,0.f,0.f,0.f,0.f};
      v8f s1 = s0;
#pragma unroll
      for (int c = 0; c < 4; c++) {
        s0 = __builtin_amdgcn_wmma_f32_16x16x32_f16(false, qf[c], false, kf0[c], (short)0, s0, false, false);
        s1 = __builtin_amdgcn_wmma_f32_16x16x32_f16(false, qf[c], false, kf1[c], (short)0, s1, false, false);
      }

      const bool diag = (kb + 31 > qb);  // wave-uniform: only 1-2 tiles need masking
      union { v8f v; float f[8]; } us0, us1; us0.v = s0; us1.v = s1;
      float corr8[8];
#pragma unroll
      for (int r = 0; r < 8; r++) {
        float a0 = us0.f[r] * scale;
        float a1 = us1.f[r] * scale;
        if (diag) {
          const int qrow = qb + r + lh * 8;
          if (kb + l16      > qrow) a0 = -1e30f;
          if (kb + 16 + l16 > qrow) a1 = -1e30f;
        }
        float mx = fmaxf(a0, a1);
#pragma unroll
        for (int off = 1; off < 16; off <<= 1) mx = fmaxf(mx, __shfl_xor(mx, off, 32));
        const float mnew = fmaxf(mrun[r], mx);
        corr8[r] = __expf(mrun[r] - mnew);
        mrun[r]  = mnew;
        const float p0 = __expf(a0 - mnew), p1 = __expf(a1 - mnew);
        const int mq = r + lh * 8;
        pbuf[wave][mq * 32 + l16]      = (_Float16)p0;
        pbuf[wave][mq * 32 + 16 + l16] = (_Float16)p1;
      }

      // rescale running output
#pragma unroll
      for (int fn = 0; fn < 8; fn++) {
        union { v8f v; float f[8]; } uo; uo.v = o[fn];
#pragma unroll
        for (int r = 0; r < 8; r++) uo.f[r] *= corr8[r];
        o[fn] = uo.v;
      }

      // P as A-fragment (LDS transpose round-trip)
      const _Float16* pp = &pbuf[wave][0];
      v16h pf = ld_32(pp + l16 * 32 + lh * 16);

      // row sums via WMMA: (P x ones)[m, *] = rowsum(m), layout matches lrun
      v8f rz = (v8f){0.f,0.f,0.f,0.f,0.f,0.f,0.f,0.f};
      v8f rsf = __builtin_amdgcn_wmma_f32_16x16x32_f16(false, pf, false, ones, (short)0, rz, false, false);
      union { v8f v; float f[8]; } urs; urs.v = rsf;
#pragma unroll
      for (int r = 0; r < 8; r++) lrun[r] = lrun[r] * corr8[r] + urs.f[r];

      // ---- O += P * V: load all 8 V fragments, then 8 independent WMMAs ----
      v16h vf[8];
#pragma unroll
      for (int fn = 0; fn < 8; fn++)
        vf[fn] = ld_32(&vtile[cur][(fn * 16 + l16) * 32 + lh * 16]);
#pragma unroll
      for (int fn = 0; fn < 8; fn++)
        o[fn] = __builtin_amdgcn_wmma_f32_16x16x32_f16(false, pf, false, vf[fn], (short)0, o[fn], false, false);
    }
    __syncthreads();  // tile reads done before next stage overwrites buffer
  }

  // normalize + store attn output [b][s][h*128+d] fp16
  float linv[8];
#pragma unroll
  for (int r = 0; r < 8; r++) linv[r] = 1.0f / lrun[r];
#pragma unroll
  for (int fn = 0; fn < 8; fn++) {
    union { v8f v; float f[8]; } uo; uo.v = o[fn];
    const int d = fn * 16 + l16;
#pragma unroll
    for (int r = 0; r < 8; r++) {
      const int row = qb + r + lh * 8;
      attn[((size_t)(b * SEQ + row)) * DIMSZ + h * HD + d] = (_Float16)(uo.f[r] * linv[r]);
    }
  }
}

// ================= host-side orchestration =================
extern "C" void kernel_launch(void* const* d_in, const int* in_sizes, int n_in,
                              void* d_out, int out_size, void* d_ws, size_t ws_size,
                              hipStream_t stream) {
  const float* x  = (const float*)d_in[0];
  const float* wq = (const float*)d_in[1];
  const float* wk = (const float*)d_in[2];
  const float* wv = (const float*)d_in[3];
  const float* wo = (const float*)d_in[4];

  char* ws = (char*)d_ws;
  size_t off = 0;
  _Float16* xb   = (_Float16*)(ws + off); off += (size_t)MROWS * DIMSZ * 2;
  _Float16* wqb  = (_Float16*)(ws + off); off += (size_t)DIMSZ * DIMSZ * 2;
  _Float16* wkb  = (_Float16*)(ws + off); off += (size_t)(NKV*HD) * DIMSZ * 2;
  _Float16* wvb  = (_Float16*)(ws + off); off += (size_t)(NKV*HD) * DIMSZ * 2;
  _Float16* wob  = (_Float16*)(ws + off); off += (size_t)DIMSZ * DIMSZ * 2;
  _Float16* Qb   = (_Float16*)(ws + off); off += (size_t)BATCH*NH*SEQ*HD * 2;
  _Float16* Kb   = (_Float16*)(ws + off); off += (size_t)BATCH*NKV*SEQ*HD * 2;
  _Float16* VTb  = (_Float16*)(ws + off); off += (size_t)BATCH*NKV*SEQ*HD * 2;
  _Float16* attn = (_Float16*)(ws + off); off += (size_t)MROWS * DIMSZ * 2;

  auto cvt = [&](const float* in, _Float16* out, int n) {
    const int n4 = n / 4;
    cvt_f32_f16v4<<<(n4 + 255) / 256, 256, 0, stream>>>(in, out, n4);
  };
  cvt(x,  xb,  MROWS * DIMSZ);
  cvt(wq, wqb, DIMSZ * DIMSZ);
  cvt(wk, wkb, NKV * HD * DIMSZ);
  cvt(wv, wvb, NKV * HD * DIMSZ);
  cvt(wo, wob, DIMSZ * DIMSZ);

  gemm_wmma<0><<<dim3(DIMSZ / 128, MROWS / 128), 256, 0, stream>>>(xb, wqb, Qb,  MROWS, DIMSZ,  DIMSZ);
  gemm_wmma<1><<<dim3((NKV*HD) / 128, MROWS / 128), 256, 0, stream>>>(xb, wkb, Kb,  MROWS, NKV*HD, DIMSZ);
  gemm_wmma<2><<<dim3((NKV*HD) / 128, MROWS / 128), 256, 0, stream>>>(xb, wvb, VTb, MROWS, NKV*HD, DIMSZ);

  flash_wmma<<<dim3(SEQ / 64, BATCH * NH), 128, 0, stream>>>(Qb, Kb, VTb, attn);

  gemm_wmma<3><<<dim3(DIMSZ / 128, MROWS / 128), 256, 0, stream>>>(attn, wob, (float*)d_out, MROWS, DIMSZ, DIMSZ);
}
